// MyChebyNet_82102594830826
// MI455X (gfx1250) — matrix operand order repaired and checked by reading, not compiled, not run on hardware
//
#include <hip/hip_runtime.h>

typedef __attribute__((ext_vector_type(16))) __bf16 v16bf;
typedef __attribute__((ext_vector_type(8)))  __bf16 v8bf;
typedef __attribute__((ext_vector_type(8)))  float  v8f;

typedef __attribute__((address_space(3))) __bf16 lds_bf16;

#define BM 128
#define BN 128
#define KT 32
#define LDA 40   // padded LDS row stride (elems) for A rows of 32 k-elems
#define LDB 40   // padded LDS row stride (elems) for B^T rows of 32 k-elems

// LDS byte offset (relative to workgroup LDS base) for async-to-LDS vdst
__device__ __forceinline__ unsigned lds_off(const __bf16* p) {
    return (unsigned)(size_t)(lds_bf16*)p;
}

// CDNA5 async copy: 16 bytes global -> LDS, tracked by ASYNCcnt (no VGPR data)
__device__ __forceinline__ void async_load_b128(unsigned ldsOff, const __bf16* g) {
    asm volatile("global_load_async_to_lds_b128 %0, %1, off"
                 :: "v"(ldsOff), "v"(g) : "memory");
}
__device__ __forceinline__ void async_wait0() {
    asm volatile("s_wait_asynccnt 0x0" ::: "memory");
}

// ---------------- graph-normalization kernels ----------------

__global__ void cheb_deg_kernel(const int* __restrict__ row,
                                const float* __restrict__ ew,
                                float* __restrict__ deg, int E) {
    int e = blockIdx.x * blockDim.x + threadIdx.x;
    if (e < E) unsafeAtomicAdd(&deg[row[e]], ew[e]);
}

__global__ void cheb_dis_kernel(float* __restrict__ deg, int N) {
    int i = blockIdx.x * blockDim.x + threadIdx.x;
    if (i < N) {
        float d = deg[i];
        deg[i] = (d > 0.0f) ? rsqrtf(d) : 0.0f;   // in place: deg -> D^{-1/2}
    }
}

__global__ void cheb_normw_kernel(const int* __restrict__ row,
                                  const int* __restrict__ col,
                                  const float* __restrict__ ew,
                                  const float* __restrict__ dis,
                                  float* __restrict__ w, int E) {
    int e = blockIdx.x * blockDim.x + threadIdx.x;
    if (e < E) w[e] = -dis[row[e]] * ew[e] * dis[col[e]];
}

// ------- SpMM: out[row] += alpha*w[e]*h[col]; (1<<logC) float4 chunks/edge ---

__global__ void cheb_spmm_kernel(const int* __restrict__ row,
                                 const int* __restrict__ col,
                                 const float* __restrict__ w,
                                 const float* __restrict__ h, int ldH,
                                 float* __restrict__ out, int ldO,
                                 int E, int logC, float alpha) {
    long t = (long)blockIdx.x * blockDim.x + threadIdx.x;
    if (t >= ((long)E << logC)) return;
    int e = (int)(t >> logC);
    int c = (int)(t & ((1 << logC) - 1)) * 4;
    float wv = alpha * w[e];
    const float4 hv = *(const float4*)(h + (size_t)col[e] * ldH + c);
    float* o = out + (size_t)row[e] * ldO + c;
    unsafeAtomicAdd(o + 0, wv * hv.x);
    unsafeAtomicAdd(o + 1, wv * hv.y);
    unsafeAtomicAdd(o + 2, wv * hv.z);
    unsafeAtomicAdd(o + 3, wv * hv.w);
}

// ---------------- small elementwise helpers ----------------

__global__ void cheb_cast4_kernel(const float4* __restrict__ src,
                                  __bf16* __restrict__ dst, long n4) {
    long i = (long)blockIdx.x * blockDim.x + threadIdx.x;
    if (i < n4) {
        float4 v = src[i];
        long b = i * 4;
        dst[b + 0] = (__bf16)v.x;
        dst[b + 1] = (__bf16)v.y;
        dst[b + 2] = (__bf16)v.z;
        dst[b + 3] = (__bf16)v.w;
    }
}

// dst[r*Dc + c] = src[r*ldS + offS + c]  (seed U with Y1 before atomic accumulate)
__global__ void cheb_copy4_kernel(const float* __restrict__ src, int ldS, int offS,
                                  float* __restrict__ dst, int Dc, int logD, long n4) {
    long i = (long)blockIdx.x * blockDim.x + threadIdx.x;
    if (i >= n4) return;
    long idx = i * 4;
    long r = idx >> logD;
    int  c = (int)(idx & (Dc - 1));
    *(float4*)(dst + r * Dc + c) = *(const float4*)(src + r * ldS + offS + c);
}

// out = Y0 - Y2 + V + bias  (+optional ReLU); Y0 at col 0, Y2 at col 2*Dc of Ycat
__global__ void cheb_combine_kernel(const float* __restrict__ Y, int ldY,
                                    int Dc, int logD,
                                    const float* __restrict__ V,
                                    const float* __restrict__ bias,
                                    float* __restrict__ outF,
                                    __bf16* __restrict__ outB,
                                    int relu, long n4) {
    long i = (long)blockIdx.x * blockDim.x + threadIdx.x;
    if (i >= n4) return;
    long idx = i * 4;
    long r = idx >> logD;
    int  c = (int)(idx & (Dc - 1));
    float4 y0 = *(const float4*)(Y + r * ldY + c);
    float4 y2 = *(const float4*)(Y + r * ldY + 2 * Dc + c);
    float4 v  = *(const float4*)(V + r * Dc + c);
    float4 b  = *(const float4*)(bias + c);
    float o0 = y0.x - y2.x + v.x + b.x;
    float o1 = y0.y - y2.y + v.y + b.y;
    float o2 = y0.z - y2.z + v.z + b.z;
    float o3 = y0.w - y2.w + v.w + b.w;
    if (relu) {
        o0 = o0 > 0.f ? o0 : 0.f;  o1 = o1 > 0.f ? o1 : 0.f;
        o2 = o2 > 0.f ? o2 : 0.f;  o3 = o3 > 0.f ? o3 : 0.f;
    }
    if (outF) {
        float4 o = {o0, o1, o2, o3};
        *(float4*)(outF + idx) = o;
    }
    if (outB) {
        outB[idx + 0] = (__bf16)o0;
        outB[idx + 1] = (__bf16)o1;
        outB[idx + 2] = (__bf16)o2;
        outB[idx + 3] = (__bf16)o3;
    }
}

// pack W[3][K][Dout] (f32) -> Bt[3*Dout][K] (bf16), pre-transposed so the GEMM
// can move B tiles with b128 transfers (no in-kernel transpose).
__global__ void cheb_packw_kernel(const float* __restrict__ W,
                                  __bf16* __restrict__ Bt,
                                  int K, int Dout, int total /* 3*K*Dout */) {
    int o = blockIdx.x * blockDim.x + threadIdx.x;
    if (o >= total) return;
    int n = o >> 8;          // K = 256: row of Bt = s*Dout + j
    int k = o & 255;
    int s = n / Dout;
    int j = n - s * Dout;
    Bt[o] = (__bf16)W[(size_t)s * K * Dout + (size_t)k * Dout + j];
}

// ---------------- bf16 WMMA GEMM: C[M,Nout] = A[M,256] @ Bt^T --------------
// A row-major [M][256]; Bt row-major [Nout][256] (pre-transposed weights).
// Block 128x128, 8 waves (4x2), wave tile 32x64, K=256.
// Double-buffered LDS fed by GLOBAL_LOAD_ASYNC_TO_LDS_B128 (ASYNCcnt), so no
// prefetch VGPRs are live across the WMMA block (no spills).

__global__ __launch_bounds__(256, 2)
void cheb_gemm_kernel(const __bf16* __restrict__ A,
                      const __bf16* __restrict__ Bt,
                      float* __restrict__ C,
                      int M, int Nout) {
    __shared__ __bf16 As[2][BM * LDA];
    __shared__ __bf16 Bs[2][BN * LDB];   // Bs[n][k]

    const int tid  = threadIdx.x;
    const int lane = tid & 31;
    const int wave = tid >> 5;
    const int wrow = wave >> 1;   // 0..3
    const int wcol = wave & 1;    // 0..1
    const int m0 = blockIdx.x * BM;
    const int n0 = blockIdx.y * BN;
    const int lm = lane & 15;
    const int kh = lane >> 4;

    v8f acc[2][4] = {};

    // per-thread tile coordinates (same pattern for A rows and Bt rows)
    const int tr0 = tid >> 2;                 // 0..63   (first chunk row)
    const int tc  = (tid & 3) * 8;            // 0,8,16,24 (k offset)

    auto loadTile = [&](int buf, int kk) {
        #pragma unroll
        for (int it = 0; it < 2; ++it) {
            int r = tr0 + it * 64;            // 0..127
            int gr = m0 + r;
            if (gr >= M) gr = M - 1;          // clamp; rows masked at store
            async_load_b128(lds_off(&As[buf][r * LDA + tc]),
                            A + (size_t)gr * 256 + kk + tc);
            async_load_b128(lds_off(&Bs[buf][r * LDB + tc]),
                            Bt + (size_t)(n0 + r) * 256 + kk + tc);
        }
    };
    auto compute = [&](int buf) {
        v16bf afrag[2], bfrag[4];
        #pragma unroll
        for (int i = 0; i < 2; ++i) {
            const __bf16* p = &As[buf][(wrow * 32 + i * 16 + lm) * LDA + kh * 8];
            union { v16bf v; v8bf h[2]; } u;
            u.h[0] = *(const v8bf*)p;
            u.h[1] = *(const v8bf*)(p + 16);
            afrag[i] = u.v;
        }
        #pragma unroll
        for (int j = 0; j < 4; ++j) {
            const __bf16* p = &Bs[buf][(wcol * 64 + j * 16 + lm) * LDB + kh * 8];
            union { v16bf v; v8bf h[2]; } u;
            u.h[0] = *(const v8bf*)p;
            u.h[1] = *(const v8bf*)(p + 16);
            bfrag[j] = u.v;
        }
        #pragma unroll
        for (int i = 0; i < 2; ++i)
            #pragma unroll
            for (int j = 0; j < 4; ++j)
                acc[i][j] = __builtin_amdgcn_wmma_f32_16x16x32_bf16(
                    false, afrag[i], false, bfrag[j],
                    (short)0, acc[i][j], false, false);
    };

    loadTile(0, 0);
    async_wait0();
    __syncthreads();
    #pragma unroll
    for (int t = 0; t < 8; ++t) {              // K = 8 * 32 = 256
        if (t < 7) loadTile((t + 1) & 1, (t + 1) * KT);  // async, overlaps WMMA
        compute(t & 1);
        if (t < 7) {
            async_wait0();                     // tile t+1 landed in LDS
            __syncthreads();
        }
    }

    // epilogue: C elem r -> row = (lane>>4)*8 + r, col = lane&15
    #pragma unroll
    for (int i = 0; i < 2; ++i) {
        #pragma unroll
        for (int j = 0; j < 4; ++j) {
            int colg = n0 + wcol * 64 + j * 16 + lm;
            #pragma unroll
            for (int r = 0; r < 8; ++r) {
                int rowg = m0 + wrow * 32 + i * 16 + kh * 8 + r;
                if (rowg < M)
                    C[(size_t)rowg * Nout + colg] = acc[i][j][r];
            }
        }
    }
}

// ---------------- launcher ----------------

extern "C" void kernel_launch(void* const* d_in, const int* in_sizes, int n_in,
                              void* d_out, int out_size, void* d_ws, size_t ws_size,
                              hipStream_t stream) {
    (void)n_in; (void)out_size; (void)ws_size;
    const float* x  = (const float*)d_in[0];
    const int*   ei = (const int*)d_in[1];
    const float* ew = (const float*)d_in[2];
    const float* W1 = (const float*)d_in[3];
    const float* b1 = (const float*)d_in[4];
    const float* W2 = (const float*)d_in[5];
    const float* b2 = (const float*)d_in[6];
    float* out = (float*)d_out;

    const long N = in_sizes[0] / 256;   // 50000 nodes
    const long E = in_sizes[2];         // 312500 edges

    char* base = (char*)d_ws;
    size_t off = 0;
    auto take = [&](size_t bytes) -> char* {
        char* p = base + off;
        off += (bytes + 255) & ~(size_t)255;
        return p;
    };
    float*  deg = (float*)take(N * 4);
    float*  wn  = (float*)take(E * 4);
    __bf16* Xb  = (__bf16*)take(N * 256 * 2);
    __bf16* Hb  = (__bf16*)take(N * 256 * 2);
    __bf16* Bt1 = (__bf16*)take((size_t)768 * 256 * 2);
    __bf16* Bt2 = (__bf16*)take((size_t)384 * 256 * 2);
    float*  Y   = (float*)take(N * 768 * 4);   // layer1 [N,768]; reused layer2 [N,384]
    float*  U   = (float*)take(N * 256 * 4);   // layer1 [N,256]; reused layer2 [N,128]
    float*  V   = (float*)take(N * 256 * 4);   // layer1 [N,256]; reused layer2 [N,128]

    const int* rowp = ei;
    const int* colp = ei + E;

    const int TB = 256;
    int gE = (int)((E + TB - 1) / TB);
    int gN = (int)((N + TB - 1) / TB);

    // ---- normalization (diag term vanishes: lambda_max = 2) ----
    hipMemsetAsync(deg, 0, N * 4, stream);
    cheb_deg_kernel<<<gE, TB, 0, stream>>>(rowp, ew, deg, (int)E);
    cheb_dis_kernel<<<gN, TB, 0, stream>>>(deg, (int)N);
    cheb_normw_kernel<<<gE, TB, 0, stream>>>(rowp, colp, ew, deg, wn, (int)E);

    // ---- casts / weight packing ----
    long xn4 = N * 256 / 4;
    cheb_cast4_kernel<<<(int)((xn4 + TB - 1) / TB), TB, 0, stream>>>(
        (const float4*)x, Xb, xn4);
    int w1t = 3 * 256 * 256, w2t = 3 * 256 * 128;
    cheb_packw_kernel<<<(w1t + TB - 1) / TB, TB, 0, stream>>>(W1, Bt1, 256, 256, w1t);
    cheb_packw_kernel<<<(w2t + TB - 1) / TB, TB, 0, stream>>>(W2, Bt2, 256, 128, w2t);

    const unsigned gm = (unsigned)((N + BM - 1) / BM);

    // ======== layer 1: Din=256, Dout=256 ========
    {
        const int Dc = 256, logD = 8, logC = 6, ldY = 768;
        dim3 g1(gm, 3 * Dc / BN);                                  // 391 x 6
        cheb_gemm_kernel<<<g1, 256, 0, stream>>>(Xb, Bt1, Y, (int)N, ldY);

        long n4 = N * Dc / 4;
        int  gC = (int)((n4 + TB - 1) / TB);
        long works = (long)E << logC;
        int  gS = (int)((works + TB - 1) / TB);

        // U = Y1 + 2*Lhat(Y2)
        cheb_copy4_kernel<<<gC, TB, 0, stream>>>(Y, ldY, Dc, U, Dc, logD, n4);
        cheb_spmm_kernel<<<gS, TB, 0, stream>>>(rowp, colp, wn,
                                                Y + 2 * Dc, ldY, U, Dc,
                                                (int)E, logC, 2.0f);
        // V = Lhat(U)
        hipMemsetAsync(V, 0, N * Dc * 4, stream);
        cheb_spmm_kernel<<<gS, TB, 0, stream>>>(rowp, colp, wn,
                                                U, Dc, V, Dc,
                                                (int)E, logC, 1.0f);
        // Hb = relu(Y0 - Y2 + V + b1)  (bf16 only; feeds layer-2 GEMM)
        cheb_combine_kernel<<<gC, TB, 0, stream>>>(Y, ldY, Dc, logD, V, b1,
                                                   nullptr, Hb, 1, n4);
    }

    // ======== layer 2: Din=256, Dout=128 ========
    {
        const int Dc = 128, logD = 7, logC = 5, ldY = 384;
        dim3 g2(gm, 3 * Dc / BN);                                  // 391 x 3
        cheb_gemm_kernel<<<g2, 256, 0, stream>>>(Hb, Bt2, Y, (int)N, ldY);

        long n4 = N * Dc / 4;
        int  gC = (int)((n4 + TB - 1) / TB);
        long works = (long)E << logC;
        int  gS = (int)((works + TB - 1) / TB);

        cheb_copy4_kernel<<<gC, TB, 0, stream>>>(Y, ldY, Dc, U, Dc, logD, n4);
        cheb_spmm_kernel<<<gS, TB, 0, stream>>>(rowp, colp, wn,
                                                Y + 2 * Dc, ldY, U, Dc,
                                                (int)E, logC, 2.0f);
        hipMemsetAsync(V, 0, N * Dc * 4, stream);
        cheb_spmm_kernel<<<gS, TB, 0, stream>>>(rowp, colp, wn,
                                                U, Dc, V, Dc,
                                                (int)E, logC, 1.0f);
        // out = Y0 - Y2 + V + b2  (fp32)
        cheb_combine_kernel<<<gC, TB, 0, stream>>>(Y, ldY, Dc, logD, V, b2,
                                                   out, nullptr, 0, n4);
    }
}
